// LoFTRModule_78864189489481
// MI455X (gfx1250) — compile-verified, matrix operand order checked by reading, not run on hardware
//
#include <hip/hip_runtime.h>
#include <hip/hip_bf16.h>
#include <stdint.h>
#include <stddef.h>

// ---------------------------------------------------------------------------
// Types for CDNA5 WMMA (wave32): V_WMMA_F32_16X16X32_BF16
// ---------------------------------------------------------------------------
typedef __attribute__((ext_vector_type(16))) __bf16 v16bf;
typedef __attribute__((ext_vector_type(8)))  float  v8f;
typedef unsigned short u16t;
typedef __attribute__((ext_vector_type(4)))  u16t  u16x4;
typedef __attribute__((ext_vector_type(8)))  u16t  u16x8;
typedef __attribute__((ext_vector_type(16))) u16t  u16x16;
typedef __attribute__((ext_vector_type(4)))  float f32x4;
typedef __attribute__((ext_vector_type(4)))  int   i32x4;

#define DEV __device__ __forceinline__

static constexpr int Bn   = 8;
static constexpr int Ln   = 4800;
static constexpr int Hn   = 8;
static constexpr int NROW = Bn * Ln;   // 38400 rows (B*L), multiple of 256
static constexpr int NLAY = 8;

// CDNA5 async global->LDS staging (ASYNCcnt path). Builtin signature (from the
// compiler diagnostic / upstream clang): (int4 addrspace(1)*, int4 addrspace(3)*,
// imm offset, imm cpol).
#if defined(__HIP_DEVICE_COMPILE__) && defined(__gfx1250__) &&                 \
    __has_builtin(__builtin_amdgcn_global_load_async_to_lds_b128) &&           \
    __has_builtin(__builtin_amdgcn_s_wait_asynccnt)
#define HAVE_ASYNC_LDS 1
#else
#define HAVE_ASYNC_LDS 0
#endif

#if HAVE_ASYNC_LDS
typedef __attribute__((address_space(1))) i32x4 as1_i32x4;
typedef __attribute__((address_space(3))) i32x4 as3_i32x4;
DEV void async_copy_b128(const void* g, void* l) {
  __builtin_amdgcn_global_load_async_to_lds_b128(
      (as1_i32x4*)g, (as3_i32x4*)l, 0, 0);
}
DEV void async_wait0() { __builtin_amdgcn_s_wait_asynccnt(0); }
#endif

DEV u16t f2bf(float f) {            // round-to-nearest-even f32 -> bf16
  uint32_t u = __float_as_uint(f);
  uint32_t r = u + 0x7FFFu + ((u >> 16) & 1u);
  return (u16t)(r >> 16);
}
DEV float bf2f(u16t h) { return __uint_as_float(((uint32_t)h) << 16); }

DEV v16bf mkfrag(u16x8 lo, u16x8 hi) {
  u16x16 t = __builtin_shufflevector(lo, hi, 0,1,2,3,4,5,6,7,8,9,10,11,12,13,14,15);
  return __builtin_bit_cast(v16bf, t);
}
DEV v8f wmma_bf16(v16bf a, v16bf b, v8f c) {
  return __builtin_amdgcn_wmma_f32_16x16x32_bf16(false, a, false, b, (short)0, c, false, false);
}

// ---------------------------------------------------------------------------
// Generic bf16 GEMM: C[M,N] = epi(A[M,K] * Bt[N,K]^T + bias) * rowmul
// Workgroup: 8 waves, tile 256(M) x 64(N); each wave owns 32x64 = 8 WMMA accums.
// A may be f32 (converted to bf16 while staging into LDS) or bf16 (async copy).
// ---------------------------------------------------------------------------
enum { EPI_BIAS_ELU1 = 0, EPI_BIAS = 1, EPI_RELU = 2 };

template<bool A_F32, int EPI>
__global__ __launch_bounds__(256) void gemm_bf16_kernel(
    const void* __restrict__ Ap, const u16t* __restrict__ Bt,
    const float* __restrict__ bias, const float* __restrict__ rowmul,
    u16t* __restrict__ Cb, int M, int K, int N)
{
  __shared__ u16t lA[256 * 40];   // row stride 40 halfwords = 80B (16B aligned)
  __shared__ u16t lB[64 * 40];
  const int tid  = threadIdx.x;
  const int wave = tid >> 5, lane = tid & 31, lr = lane & 15, hl = lane >> 4;
  const int m0 = blockIdx.y * 256, n0 = blockIdx.x * 64;

  v8f acc[8] = {};   // [mt][s]: mt in {0,1} (16-row subtiles), s in {0..3} (16-col)

  const int brow = tid >> 2, bseg = tid & 3;   // B loader: 4 threads/row, 8 k each

  for (int k0 = 0; k0 < K; k0 += 32) {
    __syncthreads();
    { // stage A tile (256 x 32): one row per thread
      u16t* dst = &lA[tid * 40];
      if (A_F32) {
        const float* ag = (const float*)Ap + (size_t)(m0 + tid) * K + k0;
        if (k0 + 32 < K) __builtin_prefetch((const void*)(ag + 32), 0, 1);
        #pragma unroll
        for (int s = 0; s < 4; ++s) {
          f32x4 f0 = *(const f32x4*)(ag + s * 8 + 0);
          f32x4 f1 = *(const f32x4*)(ag + s * 8 + 4);
          u16x8 pk;
          #pragma unroll
          for (int j = 0; j < 4; ++j) { pk[j] = f2bf(f0[j]); pk[4 + j] = f2bf(f1[j]); }
          *(u16x8*)(dst + s * 8) = pk;
        }
      } else {
        const u16t* ag = (const u16t*)Ap + (size_t)(m0 + tid) * K + k0;
        if (k0 + 32 < K) __builtin_prefetch((const void*)(ag + 32), 0, 1);
        #pragma unroll
        for (int s = 0; s < 4; ++s) {
#if HAVE_ASYNC_LDS
          async_copy_b128(ag + s * 8, dst + s * 8);
#else
          *(u16x8*)(dst + s * 8) = *(const u16x8*)(ag + s * 8);
#endif
        }
      }
    }
    { // stage B tile (64 x 32) from pre-transposed bf16 weights
      const u16t* bg = Bt + (size_t)(n0 + brow) * K + k0 + bseg * 8;
      if (k0 + 32 < K) __builtin_prefetch((const void*)(bg + 32), 0, 1);
#if HAVE_ASYNC_LDS
      async_copy_b128(bg, &lB[brow * 40 + bseg * 8]);
#else
      *(u16x8*)&lB[brow * 40 + bseg * 8] = *(const u16x8*)bg;
#endif
    }
#if HAVE_ASYNC_LDS
    async_wait0();
#endif
    __syncthreads();

    v16bf a0, a1;
    {
      const int ab0 = (wave * 32 + lr) * 40;
      const int ab1 = (wave * 32 + 16 + lr) * 40;
      a0 = mkfrag(*(const u16x8*)&lA[ab0 + hl * 8], *(const u16x8*)&lA[ab0 + 16 + hl * 8]);
      a1 = mkfrag(*(const u16x8*)&lA[ab1 + hl * 8], *(const u16x8*)&lA[ab1 + 16 + hl * 8]);
    }
    #pragma unroll
    for (int s = 0; s < 4; ++s) {
      const int bb = (s * 16 + lr) * 40;
      v16bf b = mkfrag(*(const u16x8*)&lB[bb + hl * 8],
                       *(const u16x8*)&lB[bb + 16 + hl * 8]);
      acc[s]     = wmma_bf16(a0, b, acc[s]);
      acc[4 + s] = wmma_bf16(a1, b, acc[4 + s]);
    }
  }

  // epilogue: C fragment layout -> row = r + 8*hl (within 16-row tile), col = lr
  #pragma unroll
  for (int mt = 0; mt < 2; ++mt) {
    float rm[8];
    #pragma unroll
    for (int r = 0; r < 8; ++r) {
      const int rowg = m0 + wave * 32 + mt * 16 + r + hl * 8;
      rm[r] = rowmul ? rowmul[rowg] : 1.0f;
    }
    #pragma unroll
    for (int s = 0; s < 4; ++s) {
      const int col = n0 + s * 16 + lr;
      const float bvv = bias ? bias[col] : 0.0f;
      #pragma unroll
      for (int r = 0; r < 8; ++r) {
        const int rowg = m0 + wave * 32 + mt * 16 + r + hl * 8;
        float v = acc[mt * 4 + s][r] + bvv;
        if (EPI == EPI_BIAS_ELU1) v = (v > 0.0f) ? (v + 1.0f) : __expf(v);   // elu(x)+1
        if (EPI == EPI_RELU)      v = fmaxf(v, 0.0f);
        v *= rm[r];
        Cb[(size_t)rowg * N + col] = f2bf(v);
      }
    }
  }
}

// ---------------------------------------------------------------------------
// GEMM (N fixed = 256) + optional bias + fused LayerNorm + optional residual.
// Workgroup: 32 rows x 256 cols (8 waves as 2 M-tiles x 4 N-tiles).
// ---------------------------------------------------------------------------
__global__ __launch_bounds__(256) void gemm_ln_kernel(
    const u16t* __restrict__ Ab, const u16t* __restrict__ Bt,
    const float* __restrict__ bias,
    const float* __restrict__ gamma, const float* __restrict__ beta,
    const float* __restrict__ resid, float* __restrict__ outF,
    u16t* __restrict__ outB, int M, int K)
{
  __shared__ u16t  lA[32 * 40];
  __shared__ u16t  lB[256 * 40];
  __shared__ float lC[32 * 264];
  __shared__ float lS1[32 * 8], lS2[32 * 8];
  __shared__ float lMu[32], lRs[32];

  const int tid  = threadIdx.x;
  const int wave = tid >> 5, lane = tid & 31, lr = lane & 15, hl = lane >> 4;
  const int m0 = blockIdx.x * 32;
  const int mr = (wave >> 2) * 16;       // 0 or 16
  const int nc = (wave & 3) * 64;

  v8f acc0 = {}, acc1 = {}, acc2 = {}, acc3 = {};
  const int arow = tid >> 3, aseg = tid & 7;

  for (int k0 = 0; k0 < K; k0 += 32) {
    __syncthreads();
    { // A tile 32x32
      const u16t* ag = Ab + (size_t)(m0 + arow) * K + k0 + aseg * 4;
      *(u16x4*)&lA[arow * 40 + aseg * 4] = *(const u16x4*)ag;
    }
    #pragma unroll
    for (int r0 = 0; r0 < 8; ++r0) { // B tile 256x32
      const int row = r0 * 32 + arow;
      const u16t* bg = Bt + (size_t)row * K + k0 + aseg * 4;
      *(u16x4*)&lB[row * 40 + aseg * 4] = *(const u16x4*)bg;
    }
    __syncthreads();

    const int abase = (mr + lr) * 40;
    v16bf a = mkfrag(*(const u16x8*)&lA[abase + hl * 8],
                     *(const u16x8*)&lA[abase + 16 + hl * 8]);
    #pragma unroll
    for (int s = 0; s < 4; ++s) {
      const int bb = (nc + s * 16 + lr) * 40;
      v16bf b = mkfrag(*(const u16x8*)&lB[bb + hl * 8],
                       *(const u16x8*)&lB[bb + 16 + hl * 8]);
      if (s == 0) acc0 = wmma_bf16(a, b, acc0);
      if (s == 1) acc1 = wmma_bf16(a, b, acc1);
      if (s == 2) acc2 = wmma_bf16(a, b, acc2);
      if (s == 3) acc3 = wmma_bf16(a, b, acc3);
    }
  }
  __syncthreads();

  // stash C (+bias) into LDS for the row-wise LayerNorm
  v8f accs[4] = {acc0, acc1, acc2, acc3};
  #pragma unroll
  for (int s = 0; s < 4; ++s) {
    const int col = nc + s * 16 + lr;
    const float bvv = bias ? bias[col] : 0.0f;
    #pragma unroll
    for (int r = 0; r < 8; ++r)
      lC[(mr + r + hl * 8) * 264 + col] = accs[s][r] + bvv;
  }
  __syncthreads();

  { // partial sums: thread handles (row = tid/8, 32-col segment = tid%8)
    const int row = tid >> 3, seg = tid & 7;
    float s1 = 0.f, s2 = 0.f;
    #pragma unroll
    for (int c = 0; c < 32; ++c) {
      float v = lC[row * 264 + seg * 32 + c];
      s1 += v; s2 += v * v;
    }
    lS1[row * 8 + seg] = s1; lS2[row * 8 + seg] = s2;
  }
  __syncthreads();
  if (tid < 32) {
    float s1 = 0.f, s2 = 0.f;
    #pragma unroll
    for (int j = 0; j < 8; ++j) { s1 += lS1[tid * 8 + j]; s2 += lS2[tid * 8 + j]; }
    const float mu = s1 * (1.0f / 256.0f);
    const float var = s2 * (1.0f / 256.0f) - mu * mu;
    lMu[tid] = mu; lRs[tid] = rsqrtf(var + 1e-5f);
  }
  __syncthreads();
  {
    const int row = tid >> 3, seg = tid & 7;
    const int grow = m0 + row;
    const float mu = lMu[row], rs = lRs[row];
    #pragma unroll
    for (int c = 0; c < 32; ++c) {
      const int col = seg * 32 + c;
      float v = (lC[row * 264 + col] - mu) * rs * gamma[col] + beta[col];
      if (resid) v += resid[(size_t)grow * 256 + col];
      if (outF) outF[(size_t)grow * 256 + col] = v;
      if (outB) outB[(size_t)grow * 256 + col] = f2bf(v);
    }
  }
}

// ---------------------------------------------------------------------------
// Linear-attention reduction per (b,h): P[d,m] = sum_s K[s,d]*V[s,m] (stored
// transposed as Pt[m*32+d] bf16 for the consumer B-fragment), Ksum[d] f32.
// 64 blocks x 8 waves; each wave WMMA-accumulates over strided 32-token chunks.
// ---------------------------------------------------------------------------
__global__ __launch_bounds__(256) void kv_reduce_kernel(
    const u16t* __restrict__ Kb, const u16t* __restrict__ Vb,
    u16t* __restrict__ Pt, float* __restrict__ Ksum)
{
  __shared__ __align__(16) unsigned char smem[40960];
  u16t* kAll = (u16t*)smem;                          // 8 x (32*40) halfwords
  u16t* vAll = kAll + 8 * 32 * 40;
  float (*pred)[32][33] = (float (*)[32][33])smem;   // aliases staging (after barrier)
  float* ksred = (float*)(smem + 33792);             // [8][32]

  const int tid = threadIdx.x, wave = tid >> 5, lane = tid & 31, lr = lane & 15, hl = lane >> 4;
  const int bh = blockIdx.x, b = bh >> 3, h = bh & 7;
  const u16t* Kp = Kb + (size_t)b * Ln * 256 + h * 32;
  const u16t* Vp = Vb + (size_t)b * Ln * 256 + h * 32;
  u16t* kb = kAll + wave * (32 * 40);
  u16t* vb = vAll + wave * (32 * 40);

  v8f a00 = {}, a01 = {}, a10 = {}, a11 = {};
  float ksacc = 0.0f;
  const int nch = Ln / 32;   // 150 chunks

  for (int c = wave; c < nch; c += 8) {
    const size_t srow = (size_t)(c * 32 + lane) * 256;
    #pragma unroll
    for (int j = 0; j < 4; ++j) {
      *(u16x8*)&kb[lane * 40 + j * 8] = *(const u16x8*)(Kp + srow + j * 8);
      *(u16x8*)&vb[lane * 40 + j * 8] = *(const u16x8*)(Vp + srow + j * 8);
    }
    // per-wave LDS ordering: compiler inserts dscnt waits for these dependences
    v16bf afr[2], bfr[2];
    #pragma unroll
    for (int mt = 0; mt < 2; ++mt) {   // A = K^T : A[d, s] = K[s, d]
      u16x8 lo, hi;
      #pragma unroll
      for (int j = 0; j < 8; ++j) {
        lo[j] = kb[(hl * 8 + j) * 40 + mt * 16 + lr];
        hi[j] = kb[(16 + hl * 8 + j) * 40 + mt * 16 + lr];
      }
      afr[mt] = mkfrag(lo, hi);
    }
    #pragma unroll
    for (int nt = 0; nt < 2; ++nt) {   // B = V : B[s, m]
      u16x8 lo, hi;
      #pragma unroll
      for (int j = 0; j < 8; ++j) {
        lo[j] = vb[(hl * 8 + j) * 40 + nt * 16 + lr];
        hi[j] = vb[(16 + hl * 8 + j) * 40 + nt * 16 + lr];
      }
      bfr[nt] = mkfrag(lo, hi);
    }
    a00 = wmma_bf16(afr[0], bfr[0], a00);
    a01 = wmma_bf16(afr[0], bfr[1], a01);
    a10 = wmma_bf16(afr[1], bfr[0], a10);
    a11 = wmma_bf16(afr[1], bfr[1], a11);
    #pragma unroll
    for (int s = 0; s < 32; ++s) ksacc += bf2f(kb[s * 40 + lane]);  // Ksum[d=lane]
  }
  __syncthreads();   // staging buffers dead; pred may alias them now

  #pragma unroll
  for (int mt = 0; mt < 2; ++mt)
    #pragma unroll
    for (int nt = 0; nt < 2; ++nt)
      #pragma unroll
      for (int r = 0; r < 8; ++r) {
        const int d = mt * 16 + r + hl * 8;
        const int m = nt * 16 + lr;
        float v = (mt == 0) ? (nt == 0 ? a00[r] : a01[r]) : (nt == 0 ? a10[r] : a11[r]);
        pred[wave][d][m] = v;
      }
  ksred[wave * 32 + lane] = ksacc;
  __syncthreads();

  #pragma unroll
  for (int e = 0; e < 4; ++e) {        // 1024 elements / 256 threads
    const int idx = tid * 4 + e, d = idx >> 5, m = idx & 31;
    float s = 0.f;
    #pragma unroll
    for (int w = 0; w < 8; ++w) s += pred[w][d][m];
    Pt[(size_t)bh * 1024 + m * 32 + d] = f2bf(s);   // store transposed
  }
  if (tid < 32) {
    float s = 0.f;
    #pragma unroll
    for (int w = 0; w < 8; ++w) s += ksred[w * 32 + tid];
    Ksum[bh * 32 + tid] = s;
  }
}

// Z[row,h] = 1 / (dot(Q[row,h,:], Ksum[b,h,:]) + eps)
__global__ void z_kernel(const u16t* __restrict__ Qb, const float* __restrict__ Ksum,
                         float* __restrict__ Z, int total)
{
  const int idx = blockIdx.x * 256 + threadIdx.x;
  if (idx >= total) return;
  const int h = idx & 7, row = idx >> 3, b = row / Ln;
  const u16t* q = Qb + (size_t)row * 256 + h * 32;
  const float* ks = Ksum + (b * 8 + h) * 32;
  float acc = 0.f;
  #pragma unroll
  for (int d = 0; d < 32; ++d) acc += bf2f(q[d]) * ks[d];
  Z[idx] = 1.0f / (acc + 1e-6f);
}

// attn_out[row, h*32+m] = Z[row,h] * sum_d Q[row,d] * P[d,m]  (2 WMMA per head)
__global__ __launch_bounds__(256) void attn_kernel(
    const u16t* __restrict__ Qb, const u16t* __restrict__ Pt,
    const float* __restrict__ Z, u16t* __restrict__ Ob)
{
  const int tid = threadIdx.x, wave = tid >> 5, lane = tid & 31, lr = lane & 15, hl = lane >> 4;
  const int tile = blockIdx.x * 8 + wave;
  const int tok0 = tile * 16;
  const int b = tok0 / Ln;
  for (int h = 0; h < 8; ++h) {
    const u16t* qp = Qb + (size_t)(tok0 + lr) * 256 + h * 32;
    v16bf a = mkfrag(*(const u16x8*)(qp + hl * 8), *(const u16x8*)(qp + 16 + hl * 8));
    const u16t* pp = Pt + (size_t)(b * 8 + h) * 1024;
    v8f acc[2];
    #pragma unroll
    for (int nt = 0; nt < 2; ++nt) {
      const u16t* bp = pp + (nt * 16 + lr) * 32;
      v16bf bm = mkfrag(*(const u16x8*)(bp + hl * 8), *(const u16x8*)(bp + 16 + hl * 8));
      v8f z = {};
      acc[nt] = wmma_bf16(a, bm, z);
    }
    float zr[8];
    #pragma unroll
    for (int r = 0; r < 8; ++r) zr[r] = Z[(size_t)(tok0 + r + hl * 8) * 8 + h];
    #pragma unroll
    for (int nt = 0; nt < 2; ++nt)
      #pragma unroll
      for (int r = 0; r < 8; ++r) {
        const int row = tok0 + r + hl * 8;
        const int col = h * 32 + nt * 16 + lr;
        Ob[(size_t)row * 256 + col] = f2bf(acc[nt][r] * zr[r]);
      }
  }
}

// Transpose + convert f32 weights -> bf16: Wt[c*rows + r] = bf16(W[r*cols + c])
__global__ void wconv_kernel(const float* __restrict__ W, u16t* __restrict__ Wt,
                             int rows, int cols)
{
  const int idx = blockIdx.x * 256 + threadIdx.x;
  if (idx >= rows * cols) return;
  const int r = idx / cols, c = idx % cols;
  Wt[(size_t)c * rows + r] = f2bf(W[idx]);
}

// ---------------------------------------------------------------------------
// Host orchestration
// ---------------------------------------------------------------------------
extern "C" void kernel_launch(void* const* d_in, const int* in_sizes, int n_in,
                              void* d_out, int out_size, void* d_ws, size_t ws_size,
                              hipStream_t stream)
{
  (void)in_sizes; (void)n_in; (void)out_size; (void)ws_size;
  const float* desc0 = (const float*)d_in[0];
  const float* desc1 = (const float*)d_in[1];
  const float* mask0 = (const float*)d_in[2];
  const float* mask1 = (const float*)d_in[3];
  const float* Wq = (const float*)d_in[4];  const float* bq = (const float*)d_in[5];
  const float* Wk = (const float*)d_in[6];  const float* bk = (const float*)d_in[7];
  const float* Wv = (const float*)d_in[8];  const float* bv = (const float*)d_in[9];
  const float* Wm = (const float*)d_in[10]; const float* bm = (const float*)d_in[11];
  const float* w1 = (const float*)d_in[12]; const float* w2 = (const float*)d_in[13];
  const float* g1 = (const float*)d_in[14]; const float* b1 = (const float*)d_in[15];
  const float* g2 = (const float*)d_in[16]; const float* b2 = (const float*)d_in[17];

  uint8_t* p = (uint8_t*)d_ws;
  auto carve = [&](size_t bytes) -> void* {
    void* r = (void*)p; p += (bytes + 255) & ~(size_t)255; return r;
  };
  const size_t SZ_W256 = (size_t)256 * 256 * sizeof(u16t);
  const size_t SZ_WMLP = (size_t)NLAY * 512 * 256 * sizeof(u16t);
  const size_t SZ_XF   = (size_t)NROW * 256 * sizeof(float);
  const size_t SZ_XB   = (size_t)NROW * 256 * sizeof(u16t);
  const size_t SZ_HB   = (size_t)NROW * 512 * sizeof(u16t);

  u16t* WqT = (u16t*)carve(SZ_W256);
  u16t* WkT = (u16t*)carve(SZ_W256);
  u16t* WvT = (u16t*)carve(SZ_W256);
  u16t* WmT = (u16t*)carve(SZ_W256);
  u16t* w1T = (u16t*)carve(SZ_WMLP);       // [NLAY][512][256]
  u16t* w2T = (u16t*)carve(SZ_WMLP);       // [NLAY][256][512]
  float* x0f = (float*)carve(SZ_XF);
  float* x1f = (float*)carve(SZ_XF);
  u16t* Qb  = (u16t*)carve(SZ_XB);
  u16t* Kb  = (u16t*)carve(SZ_XB);
  u16t* Vb  = (u16t*)carve(SZ_XB);
  u16t* Ab  = (u16t*)carve(SZ_XB);         // attention output (bf16)
  u16t* Mb  = (u16t*)carve(SZ_XB);         // layernormed message (bf16)
  u16t* Hb  = (u16t*)carve(SZ_HB);         // MLP hidden (bf16)
  u16t* PtB = (u16t*)carve((size_t)64 * 1024 * sizeof(u16t));
  float* Ks = (float*)carve((size_t)64 * 32 * sizeof(float));
  float* Zb = (float*)carve((size_t)NROW * 8 * sizeof(float));

  // one-time (per call) weight transpose+convert
  wconv_kernel<<<(256 * 256 + 255) / 256, 256, 0, stream>>>(Wq, WqT, 256, 256);
  wconv_kernel<<<(256 * 256 + 255) / 256, 256, 0, stream>>>(Wk, WkT, 256, 256);
  wconv_kernel<<<(256 * 256 + 255) / 256, 256, 0, stream>>>(Wv, WvT, 256, 256);
  wconv_kernel<<<(256 * 256 + 255) / 256, 256, 0, stream>>>(Wm, WmT, 256, 256);
  for (int l = 0; l < NLAY; ++l) {
    wconv_kernel<<<(256 * 512 + 255) / 256, 256, 0, stream>>>(
        w1 + (size_t)l * 256 * 512, w1T + (size_t)l * 512 * 256, 256, 512);
    wconv_kernel<<<(512 * 256 + 255) / 256, 256, 0, stream>>>(
        w2 + (size_t)l * 512 * 256, w2T + (size_t)l * 256 * 512, 512, 256);
  }
  (void)hipMemcpyAsync(x0f, desc0, SZ_XF, hipMemcpyDeviceToDevice, stream);
  (void)hipMemcpyAsync(x1f, desc1, SZ_XF, hipMemcpyDeviceToDevice, stream);

  auto half_layer = [&](float* x, const float* src, const float* xm, const float* sm, int l) {
    dim3 gp(4, NROW / 256);   // 256/64 col-blocks x 150 row-blocks
    gemm_bf16_kernel<true, EPI_BIAS_ELU1><<<gp, 256, 0, stream>>>(x,   WqT, bq, xm, Qb, NROW, 256, 256);
    gemm_bf16_kernel<true, EPI_BIAS_ELU1><<<gp, 256, 0, stream>>>(src, WkT, bk, sm, Kb, NROW, 256, 256);
    gemm_bf16_kernel<true, EPI_BIAS>     <<<gp, 256, 0, stream>>>(src, WvT, bv, sm, Vb, NROW, 256, 256);
    kv_reduce_kernel<<<64, 256, 0, stream>>>(Kb, Vb, PtB, Ks);
    z_kernel<<<(NROW * 8 + 255) / 256, 256, 0, stream>>>(Qb, Ks, Zb, NROW * 8);
    attn_kernel<<<NROW / 128, 256, 0, stream>>>(Qb, PtB, Zb, Ab);
    // merge + LN(g1,b1) -> Mb (bf16)
    gemm_ln_kernel<<<NROW / 32, 256, 0, stream>>>(Ab, WmT, bm, g1 + l * 256, b1 + l * 256,
                                                  nullptr, nullptr, Mb, NROW, 256);
    // MLP1: relu(Mb @ w1) -> Hb
    dim3 g1d(8, NROW / 256);
    gemm_bf16_kernel<false, EPI_RELU><<<g1d, 256, 0, stream>>>(
        Mb, w1T + (size_t)l * 512 * 256, nullptr, nullptr, Hb, NROW, 256, 512);
    // MLP2 + LN(g2,b2) + residual: x = x + LN(Hb @ w2)   (in-place elementwise)
    gemm_ln_kernel<<<NROW / 32, 256, 0, stream>>>(Hb, w2T + (size_t)l * 256 * 512, nullptr,
                                                  g2 + l * 256, b2 + l * 256, x, x, nullptr,
                                                  NROW, 512);
  };

  for (int i = 0; i < NLAY; ++i) {
    const bool self = ((i & 1) == 0);   // self, cross, self, cross, ...
    if (self) {
      half_layer(x0f, x0f, mask0, mask0, i);
      half_layer(x1f, x1f, mask1, mask1, i);
    } else {
      half_layer(x0f, x1f, mask0, mask1, i);   // desc0 first,
      half_layer(x1f, x0f, mask1, mask0, i);   // desc1 sees updated desc0
    }
  }

  float* out = (float*)d_out;
  (void)hipMemcpyAsync(out, x0f, SZ_XF, hipMemcpyDeviceToDevice, stream);
  (void)hipMemcpyAsync(out + (size_t)NROW * 256, x1f, SZ_XF, hipMemcpyDeviceToDevice, stream);
}
